// EGAMultiHeadedAttention_14405320310792
// MI455X (gfx1250) — compile-verified, matrix operand order checked by reading, not run on hardware
//
#include <hip/hip_runtime.h>
#include <hip/hip_bf16.h>
#include <math.h>

// ---------------------------------------------------------------------------
// Problem dims (fixed by setup_inputs)
// ---------------------------------------------------------------------------
constexpr int Bb = 8, Ls = 2048, Ss = 2048, Ee = 1024;
constexpr int Hh = 16, Gg = 64;
constexpr int KDim = 1024, VDim = 1024, HG = 1024;
constexpr int Mrows = Bb * Ls;              // 16384
constexpr float SCALE = 0.125f;             // 1/sqrt(64)
constexpr float NEGV = -1e9f;

typedef _Float16 half16 __attribute__((ext_vector_type(16)));
typedef _Float16 half8  __attribute__((ext_vector_type(8)));
typedef float    v8f    __attribute__((ext_vector_type(8)));
typedef float    f32x4  __attribute__((ext_vector_type(4)));

// ---------------------------------------------------------------------------
// WMMA GEMM: C[MxN] = A[MxK] * B[KxN], all fp32 row-major in memory,
// f16 multiply / f32 accumulate via v_wmma_f32_16x16x32_f16.
// Block tile 128x128, K-tile 32, 256 threads = 8 waves, wave tile 32x64.
// N, K are compile-time so store/load strides fold into instruction offsets.
// ---------------------------------------------------------------------------
constexpr int BM = 128, BN = 128, BK = 32;
constexpr int SAK = BK + 8;   // LDS row stride (halves) for A, 80B = 16B-aligned rows
constexpr int SBK = BK + 8;   // LDS column stride (halves) for B (stored col-major)

union AF { half16 v; half8 h[2]; };

template <int N, int K>
__global__ __launch_bounds__(256)
void gemm_wmma_f16(const float* __restrict__ A, const float* __restrict__ Bm,
                   float* __restrict__ C) {
  __shared__ _Float16 sA[BM * SAK];
  __shared__ _Float16 sB[BN * SBK];

  const int tid  = threadIdx.x;
  const int lane = tid & 31;
  const int wid  = tid >> 5;
  const int wrow = (wid & 3) * 32;     // wave's row offset in block tile
  const int wcol = (wid >> 2) * 64;    // wave's col offset in block tile
  const int lr   = lane & 15;          // row (A) / col (B) within 16
  const int lko  = (lane >> 4) << 3;   // K sub-offset: 0 or 8 (ISA 16-bit A layout)

  const int m0 = blockIdx.y * BM;
  const int n0 = blockIdx.x * BN;

  // A loader: thread -> (row, 16-wide K segment)
  const int ar = tid >> 1;
  const int ac = (tid & 1) * 16;
  // B loader: thread -> (column, 16-deep K segment)
  const int bc = tid & 127;
  const int bk = (tid >> 7) * 16;

  const float* gaBase = A + (size_t)(m0 + ar) * K + ac;
  const float* gbBase = Bm + (size_t)bk * N + n0 + bc;

  v8f acc[2][4];
#pragma unroll
  for (int i = 0; i < 2; ++i)
#pragma unroll
    for (int j = 0; j < 4; ++j) acc[i][j] = 0.f;

  for (int kt = 0; kt < K; kt += BK) {
    __syncthreads();
    {   // stage A tile (row-major), fp32 -> f16
      const float* ga = gaBase + kt;
      f32x4 a0 = *(const f32x4*)(ga + 0);
      f32x4 a1 = *(const f32x4*)(ga + 4);
      f32x4 a2 = *(const f32x4*)(ga + 8);
      f32x4 a3 = *(const f32x4*)(ga + 12);
      half8 h0, h1;
#pragma unroll
      for (int j = 0; j < 4; ++j) { h0[j] = (_Float16)a0[j]; h0[4 + j] = (_Float16)a1[j]; }
#pragma unroll
      for (int j = 0; j < 4; ++j) { h1[j] = (_Float16)a2[j]; h1[4 + j] = (_Float16)a3[j]; }
      *(half8*)&sA[ar * SAK + ac]     = h0;
      *(half8*)&sA[ar * SAK + ac + 8] = h1;
    }
    {   // stage B tile transposed to column-major so the B fragment is contiguous
      const float* gb = gbBase + (size_t)kt * N;
      half8 h0, h1;
#pragma unroll
      for (int j = 0; j < 8; ++j) h0[j] = (_Float16)gb[(size_t)j * N];
#pragma unroll
      for (int j = 0; j < 8; ++j) h1[j] = (_Float16)gb[(size_t)(8 + j) * N];
      *(half8*)&sB[bc * SBK + bk]     = h0;
      *(half8*)&sB[bc * SBK + bk + 8] = h1;
    }
    __syncthreads();

    // Fragments per ISA 16-bit 16x32 layout:
    //  halves 0..7  = K[lko .. lko+7],  halves 8..15 = K[16+lko .. 16+lko+7]
    AF af[2], bf[4];
#pragma unroll
    for (int rt = 0; rt < 2; ++rt) {
      const int base = (wrow + rt * 16 + lr) * SAK + lko;
      af[rt].h[0] = *(const half8*)&sA[base];
      af[rt].h[1] = *(const half8*)&sA[base + 16];
    }
#pragma unroll
    for (int ct = 0; ct < 4; ++ct) {
      const int base = (wcol + ct * 16 + lr) * SBK + lko;
      bf[ct].h[0] = *(const half8*)&sB[base];
      bf[ct].h[1] = *(const half8*)&sB[base + 16];
    }
#pragma unroll
    for (int rt = 0; rt < 2; ++rt)
#pragma unroll
      for (int ct = 0; ct < 4; ++ct) {
#if defined(__gfx1250__)
        acc[rt][ct] = __builtin_amdgcn_wmma_f32_16x16x32_f16(
            false, af[rt].v, false, bf[ct].v, (short)0, acc[rt][ct], false, false);
#endif
      }
  }

  // C/D layout: VGPR v -> row v (+8 for lanes 16..31), col = lane&15.
  // N is constexpr: the 8 row stores use constant instruction offsets (v*N*4B).
#pragma unroll
  for (int rt = 0; rt < 2; ++rt)
#pragma unroll
    for (int ct = 0; ct < 4; ++ct) {
      const int col   = n0 + wcol + ct * 16 + lr;
      const int rbase = m0 + wrow + rt * 16 + ((lane >> 4) << 3);
      float* cptr = C + (size_t)rbase * N + col;
#pragma unroll
      for (int v = 0; v < 8; ++v)
        cptr[(size_t)v * N] = acc[rt][ct][v];
    }
}

// ---------------------------------------------------------------------------
// K2: importance softmax over L + gqs = sum_l softmax(ip)[l] * qs[l,:]
// One block per (b,h,g), 256 threads.
// ---------------------------------------------------------------------------
__global__ __launch_bounds__(256)
void importance_gq(const float* __restrict__ IP, const float* __restrict__ QP,
                   float* __restrict__ GQ) {
  const int tid = threadIdx.x;
  const int bid = blockIdx.x;
  const int g = bid % Gg;
  const int h = (bid / Gg) % Hh;
  const int b = bid / (Gg * Hh);

  __shared__ float red[256];
  __shared__ float partial[4][64];
  __shared__ float bmax, bsum;

  const float* ipcol = IP + (size_t)b * Ls * HG + h * Gg + g;   // [l] -> ipcol[l*HG]

  float m = -INFINITY;
  for (int l = tid; l < Ls; l += 256) m = fmaxf(m, ipcol[(size_t)l * HG]);
  red[tid] = m; __syncthreads();
  for (int s = 128; s > 0; s >>= 1) { if (tid < s) red[tid] = fmaxf(red[tid], red[tid + s]); __syncthreads(); }
  if (tid == 0) bmax = red[0];
  __syncthreads();
  m = bmax;

  float sum = 0.f;
  for (int l = tid; l < Ls; l += 256) sum += __expf(ipcol[(size_t)l * HG] - m);
  red[tid] = sum; __syncthreads();
  for (int s = 128; s > 0; s >>= 1) { if (tid < s) red[tid] += red[tid + s]; __syncthreads(); }
  if (tid == 0) bsum = red[0];
  __syncthreads();
  const float inv = 1.f / bsum;

  const int e = tid & 63, ch = tid >> 6;
  const float* qcol = QP + (size_t)b * Ls * KDim + h * 64 + e;
  float acc = 0.f;
  for (int l = ch; l < Ls; l += 4) {
    const float w = __expf(ipcol[(size_t)l * HG] - m) * inv;
    acc += w * qcol[(size_t)l * KDim];
  }
  partial[ch][e] = acc;
  __syncthreads();
  if (tid < 64)
    GQ[(size_t)bid * 64 + tid] =
        partial[0][tid] + partial[1][tid] + partial[2][tid] + partial[3][tid];
}

// ---------------------------------------------------------------------------
// K3: g_scores = gq . kk^T * scale, mask, softmax over S,
//     global_keys = attn@kk, global_values = attn@vs.
// One block per (b,h,g), 256 threads, scores resident in LDS.
// ---------------------------------------------------------------------------
__global__ __launch_bounds__(256)
void stage1_global(const float* __restrict__ GQ, const float* __restrict__ KP,
                   const float* __restrict__ VP, const unsigned char* __restrict__ mask,
                   float* __restrict__ GK, float* __restrict__ GV) {
  const int tid = threadIdx.x;
  const int bid = blockIdx.x;
  const int h = (bid / Gg) % Hh;
  const int b = bid / (Gg * Hh);

  __shared__ float gq[64];
  __shared__ float sc[Ss];
  __shared__ float red[256];
  __shared__ float pk[4][64], pv[4][64];
  __shared__ float bmax, bsum;

  if (tid < 64) gq[tid] = GQ[(size_t)bid * 64 + tid];
  __syncthreads();

  for (int s = tid; s < Ss; s += 256) {
    const float* krow = KP + ((size_t)b * Ss + s) * KDim + h * 64;
    float d = 0.f;
#pragma unroll
    for (int e = 0; e < 64; ++e) d += gq[e] * krow[e];
    d *= SCALE;
    if (mask[(size_t)b * Ss + s]) d = NEGV;
    sc[s] = d;
  }
  __syncthreads();

  float m = -INFINITY;
  for (int s = tid; s < Ss; s += 256) m = fmaxf(m, sc[s]);
  red[tid] = m; __syncthreads();
  for (int st = 128; st > 0; st >>= 1) { if (tid < st) red[tid] = fmaxf(red[tid], red[tid + st]); __syncthreads(); }
  if (tid == 0) bmax = red[0];
  __syncthreads();
  m = bmax;

  float sum = 0.f;
  for (int s = tid; s < Ss; s += 256) { const float e = __expf(sc[s] - m); sc[s] = e; sum += e; }
  __syncthreads();
  red[tid] = sum; __syncthreads();
  for (int st = 128; st > 0; st >>= 1) { if (tid < st) red[tid] += red[tid + st]; __syncthreads(); }
  if (tid == 0) bsum = red[0];
  __syncthreads();
  const float inv = 1.f / bsum;

  const int e = tid & 63, ch = tid >> 6;
  const float* kcol = KP + (size_t)b * Ss * KDim + h * 64 + e;
  const float* vcol = VP + (size_t)b * Ss * VDim + h * 64 + e;
  float ak = 0.f, av = 0.f;
  for (int s = ch; s < Ss; s += 4) {
    const float a = sc[s] * inv;
    ak += a * kcol[(size_t)s * KDim];
    av += a * vcol[(size_t)s * VDim];
  }
  pk[ch][e] = ak; pv[ch][e] = av;
  __syncthreads();
  if (tid < 64) {
    GK[(size_t)bid * 64 + tid] = pk[0][tid] + pk[1][tid] + pk[2][tid] + pk[3][tid];
    GV[(size_t)bid * 64 + tid] = pv[0][tid] + pv[1][tid] + pv[2][tid] + pv[3][tid];
  }
}

// ---------------------------------------------------------------------------
// K4: per (b,h,l): l_scores over G, softmax (emitted as l_attn output),
//     out_head = l_attn @ global_values, stored (B,L,VD) for final GEMM.
// One block per (b,h,l), 64 threads.
// ---------------------------------------------------------------------------
__global__ __launch_bounds__(64)
void stage2_attn(const float* __restrict__ QP, const float* __restrict__ GK,
                 const float* __restrict__ GV, float* __restrict__ lattn,
                 float* __restrict__ OH) {
  const int t = threadIdx.x;
  const int bid = blockIdx.x;
  const int l = bid % Ls;
  const int h = (bid / Ls) % Hh;
  const int b = bid / (Ls * Hh);

  __shared__ float lq[64], sattn[64], red[64];
  __shared__ float bmax, bsum;

  lq[t] = QP[((size_t)b * Ls + l) * KDim + h * 64 + t];
  __syncthreads();

  const float* gk = GK + (((size_t)(b * Hh + h)) * Gg + t) * 64;
  float sc = 0.f;
#pragma unroll
  for (int e = 0; e < 64; ++e) sc += lq[e] * gk[e];
  sc *= SCALE;

  red[t] = sc; __syncthreads();
  for (int s = 32; s > 0; s >>= 1) { if (t < s) red[t] = fmaxf(red[t], red[t + s]); __syncthreads(); }
  if (t == 0) bmax = red[0];
  __syncthreads();
  const float ex = __expf(sc - bmax);
  red[t] = ex; __syncthreads();
  for (int s = 32; s > 0; s >>= 1) { if (t < s) red[t] += red[t + s]; __syncthreads(); }
  if (t == 0) bsum = red[0];
  __syncthreads();

  const float a = ex / bsum;
  lattn[(((size_t)(b * Hh + h)) * Ls + l) * Gg + t] = a;   // output #2: (B,H,L,G)
  sattn[t] = a;
  __syncthreads();

  const float* gv = GV + ((size_t)(b * Hh + h)) * Gg * 64 + t;
  float o = 0.f;
#pragma unroll 8
  for (int g = 0; g < Gg; ++g) o += sattn[g] * gv[(size_t)g * 64];
  OH[((size_t)b * Ls + l) * VDim + h * 64 + t] = o;
}

// ---------------------------------------------------------------------------
// Host launcher
// ---------------------------------------------------------------------------
extern "C" void kernel_launch(void* const* d_in, const int* in_sizes, int n_in,
                              void* d_out, int out_size, void* d_ws, size_t ws_size,
                              hipStream_t stream) {
  (void)in_sizes; (void)n_in; (void)out_size; (void)ws_size;

  const float* q  = (const float*)d_in[0];
  const float* k  = (const float*)d_in[1];
  const float* v  = (const float*)d_in[2];
  const unsigned char* mask = (const unsigned char*)d_in[4];   // jnp.bool_
  const float* wq = (const float*)d_in[6];
  const float* wk = (const float*)d_in[7];
  const float* wv = (const float*)d_in[8];
  const float* wi = (const float*)d_in[9];
  const float* wo = (const float*)d_in[10];

  float* out   = (float*)d_out;
  float* lattn = out + (size_t)Mrows * Ee;   // outputs concatenated: out, l_attn

  float* ws = (float*)d_ws;
  size_t o = 0;
  float* QP = ws + o; o += (size_t)Mrows * KDim;
  float* KPp = ws + o; o += (size_t)Mrows * KDim;
  float* VPp = ws + o; o += (size_t)Mrows * VDim;
  float* IP = ws + o; o += (size_t)Mrows * HG;
  float* GQ = ws + o; o += (size_t)Bb * Hh * Gg * 64;
  float* GK = ws + o; o += (size_t)Bb * Hh * Gg * 64;
  float* GV = ws + o; o += (size_t)Bb * Hh * Gg * 64;
  float* OH = IP;   // IP is dead after importance_gq; reuse for per-head output

  const dim3 blk(256);
  const dim3 gProj(1024 / BN, Mrows / BM);   // (8, 128)

  gemm_wmma_f16<1024, 1024><<<gProj, blk, 0, stream>>>(q, wq, QP);
  gemm_wmma_f16<1024, 1024><<<gProj, blk, 0, stream>>>(k, wk, KPp);
  gemm_wmma_f16<1024, 1024><<<gProj, blk, 0, stream>>>(v, wv, VPp);
  gemm_wmma_f16<1024, 1024><<<gProj, blk, 0, stream>>>(q, wi, IP);

  importance_gq<<<Bb * Hh * Gg, blk, 0, stream>>>(IP, QP, GQ);
  stage1_global<<<Bb * Hh * Gg, blk, 0, stream>>>(GQ, KPp, VPp, mask, GK, GV);
  stage2_attn<<<Bb * Hh * Ls, dim3(64), 0, stream>>>(QP, GK, GV, lattn, OH);

  gemm_wmma_f16<1024, 1024><<<gProj, blk, 0, stream>>>(OH, wo, out);
}
